// HeteroGraphormerHeteroLite_38276748542415
// MI455X (gfx1250) — matrix operand order repaired and checked by reading, not compiled, not run on hardware
//
#include <hip/hip_runtime.h>
#include <math.h>
#include <stdint.h>
#include <stddef.h>

#define CCH 128
#define HH 4
#define DHH 32
#define MAXB 8
#define LN_EPS 1e-5f

typedef __attribute__((ext_vector_type(16))) __bf16 v16bf;
typedef __attribute__((ext_vector_type(8)))  __bf16 v8bf;
typedef __attribute__((ext_vector_type(8)))  float  v8f;

// ---------------- device helpers ----------------

__device__ __forceinline__ int bucket_of(unsigned d) {
  float b = floorf(log2f((float)d + 1.0f));
  int ib = (int)b;
  return ib < 0 ? 0 : (ib > MAXB ? MAXB : ib);
}

// order-preserving float<->uint encoding for atomicMax-based segment max
__device__ __forceinline__ unsigned fenc(float f) {
  unsigned u = __float_as_uint(f);
  return (u & 0x80000000u) ? ~u : (u | 0x80000000u);
}
__device__ __forceinline__ float fdec(unsigned u) {
  unsigned v = (u & 0x80000000u) ? (u & 0x7FFFFFFFu) : ~u;
  return __uint_as_float(v);
}

// ---------------- utility kernels ----------------

__global__ void zero_u32_kernel(unsigned* p, long n) {
  long t = (long)blockIdx.x * blockDim.x + threadIdx.x;
  if (t < n) p[t] = 0u;
}

__global__ void degree_kernel(const int* __restrict__ ei, int E,
                              unsigned* __restrict__ outDegSrc,
                              unsigned* __restrict__ inDegDst) {
  int t = blockIdx.x * blockDim.x + threadIdx.x;
  if (t >= E) return;
  atomicAdd(&outDegSrc[ei[t]], 1u);
  atomicAdd(&inDegDst[ei[E + t]], 1u);
}

// W [K, Nout] fp32 (row-major) -> WT [Nout, K] bf16 (row-major), i.e. B^T for WMMA B loads
__global__ void wcvt_kernel(const float* __restrict__ W, __bf16* __restrict__ WT,
                            int K, int Nout) {
  int t = blockIdx.x * blockDim.x + threadIdx.x;
  if (t >= K * Nout) return;
  int k = t / Nout, n = t - k * Nout;
  WT[(long)n * K + k] = (__bf16)W[t];
}

__global__ void cvt_bf16_kernel(const float* __restrict__ in, __bf16* __restrict__ out, long n) {
  long t = (long)blockIdx.x * blockDim.x + threadIdx.x;
  if (t < n) out[t] = (__bf16)in[t];
}

// ---------------- embed + LayerNorm ----------------

// one block (128 threads) per node; adds degree-bucket embeddings then LN1, emits bf16
__global__ void embed_ln_kernel(const float* __restrict__ x,
                                const unsigned* __restrict__ inDeg,
                                const unsigned* __restrict__ outDeg,
                                const float* __restrict__ in_emb,
                                const float* __restrict__ out_emb,
                                const float* __restrict__ g, const float* __restrict__ b,
                                __bf16* __restrict__ out) {
  int n = blockIdx.x, c = threadIdx.x;
  __shared__ float red[CCH];
  int bi = bucket_of(inDeg[n]);
  int bo = bucket_of(outDeg[n]);
  float v = x[(long)n * CCH + c] + in_emb[bi * CCH + c] + out_emb[bo * CCH + c];
  red[c] = v; __syncthreads();
  for (int off = CCH / 2; off > 0; off >>= 1) { if (c < off) red[c] += red[c + off]; __syncthreads(); }
  float mean = red[0] * (1.0f / CCH); __syncthreads();
  float dv = v - mean;
  red[c] = dv * dv; __syncthreads();
  for (int off = CCH / 2; off > 0; off >>= 1) { if (c < off) red[c] += red[c + off]; __syncthreads(); }
  float var = red[0] * (1.0f / CCH);
  out[(long)n * CCH + c] = (__bf16)(dv * rsqrtf(var + LN_EPS) * g[c] + b[c]);
}

// plain LN (fp32 in, bf16 out)
__global__ void ln_kernel(const float* __restrict__ x,
                          const float* __restrict__ g, const float* __restrict__ b,
                          __bf16* __restrict__ out) {
  int n = blockIdx.x, c = threadIdx.x;
  __shared__ float red[CCH];
  float v = x[(long)n * CCH + c];
  red[c] = v; __syncthreads();
  for (int off = CCH / 2; off > 0; off >>= 1) { if (c < off) red[c] += red[c + off]; __syncthreads(); }
  float mean = red[0] * (1.0f / CCH); __syncthreads();
  float dv = v - mean;
  red[c] = dv * dv; __syncthreads();
  for (int off = CCH / 2; off > 0; off >>= 1) { if (c < off) red[c] += red[c + off]; __syncthreads(); }
  float var = red[0] * (1.0f / CCH);
  out[(long)n * CCH + c] = (__bf16)(dv * rsqrtf(var + LN_EPS) * g[c] + b[c]);
}

// ---------------- WMMA GEMM ----------------
// C[M,Nout] = A[M,K](bf16) @ BT[Nout,K](bf16)^T + epilogue
// modes: 0: outF = acc + bias
//        1: outB = bf16(gelu(acc + bias))          (exact erf gelu)
//        2: outF = resid + acc + bias*rowCnt[row]  (attention out + residual, per-edge bo)
//        3: outF = resid + acc + bias              (FFN out + residual)
// One wave computes one 16x16 tile; block = 8 waves covering 128 output columns.
__global__ void gemm16_bf16(const __bf16* __restrict__ A, const __bf16* __restrict__ BT,
                            const float* __restrict__ bias,
                            const float* __restrict__ resid,
                            const unsigned* __restrict__ rowCnt,
                            float* __restrict__ outF, __bf16* __restrict__ outB,
                            int M, int K, int Nout, int mode) {
  int wave  = threadIdx.x >> 5;
  int lane  = threadIdx.x & 31;
  int lhalf = lane >> 4;     // which 16-lane half
  int lrow  = lane & 15;
  int tm = blockIdx.x * 16;
  int tn = (blockIdx.y * 8 + wave) * 16;

  const __bf16* aPtr = A  + (long)(tm + lrow) * K;  // A row for this lane (M = lrow)
  const __bf16* bPtr = BT + (long)(tn + lrow) * K;  // B column (stored transposed)

  v8f acc = {};
  for (int k0 = 0; k0 < K; k0 += 32) {
    __builtin_prefetch(aPtr + k0 + 64, 0, 1);   // global_prefetch
    __builtin_prefetch(bPtr + k0 + 64, 0, 1);
    // A fragment per ISA 16-bit A 16x32 layout: lanes0-15 K=0..7 & 16..23, lanes16-31 K=8..15 & 24..31
    v8bf a0 = *(const v8bf*)(aPtr + k0 + lhalf * 8);
    v8bf a1 = *(const v8bf*)(aPtr + k0 + 16 + lhalf * 8);
    // B fragment: same K order as A: lanes0-15 K=0..7 & 16..23, lanes16-31 K=8..15 & 24..31
    v8bf b0 = *(const v8bf*)(bPtr + k0 + lhalf * 8);
    v8bf b1 = *(const v8bf*)(bPtr + k0 + 16 + lhalf * 8);
    v16bf a, b;
#pragma unroll
    for (int i = 0; i < 8; ++i) { a[i] = a0[i]; a[i + 8] = a1[i]; b[i] = b0[i]; b[i + 8] = b1[i]; }
    acc = __builtin_amdgcn_wmma_f32_16x16x32_bf16(false, a, false, b, (short)0, acc, false, false);
  }

  int col = tn + lrow;
  float bcol = bias[col];
#pragma unroll
  for (int i = 0; i < 8; ++i) {
    int row = tm + i + lhalf * 8;          // C/D layout: VGPR i -> M = i (+8 for upper lanes)
    long idx = (long)row * Nout + col;
    float v = acc[i] + ((mode == 2) ? bcol * (float)rowCnt[row] : bcol);
    if (mode == 0) {
      outF[idx] = v;
    } else if (mode == 1) {
      outB[idx] = (__bf16)(0.5f * v * (1.0f + erff(v * 0.70710678118654752f)));
    } else {
      outF[idx] = resid[idx] + v;
    }
  }
}

// ---------------- edge attention ----------------

__global__ void edge_score_kernel(const int* __restrict__ ei, int E,
                                  const float* __restrict__ qDst, const float* __restrict__ kSrc,
                                  const unsigned* __restrict__ outDegS,
                                  const unsigned* __restrict__ inDegD,
                                  const float* __restrict__ rel, const float* __restrict__ tp,
                                  const float* __restrict__ alpha, const float* __restrict__ beta,
                                  float* __restrict__ scores, unsigned* __restrict__ segMax) {
  long t = (long)blockIdx.x * blockDim.x + threadIdx.x;
  if (t >= (long)E * HH) return;
  int e = (int)(t >> 2), h = (int)(t & 3);
  int src = ei[e], dst = ei[E + e];
  const float4* q = (const float4*)(qDst + (long)dst * CCH + h * DHH);
  const float4* k = (const float4*)(kSrc + (long)src * CCH + h * DHH);
  float acc = 0.0f;
#pragma unroll
  for (int i = 0; i < DHH / 4; ++i) {
    float4 a = q[i], b = k[i];
    acc += a.x * b.x + a.y * b.y + a.z * b.z + a.w * b.w;
  }
  float s = acc * 0.17677669529663687f   // 1/sqrt(32)
          + rel[h] + tp[h]
          + log1pf((float)outDegS[src]) * alpha[h]
          + log1pf((float)inDegD[dst]) * beta[h];
  scores[t] = s;
  atomicMax(&segMax[dst * HH + h], fenc(s));
}

__global__ void edge_exp_kernel(const int* __restrict__ ei, int E,
                                float* __restrict__ scores,
                                const unsigned* __restrict__ segMax,
                                float* __restrict__ segSum) {
  long t = (long)blockIdx.x * blockDim.x + threadIdx.x;
  if (t >= (long)E * HH) return;
  int e = (int)(t >> 2), h = (int)(t & 3);
  int dst = ei[E + e];
  float m = fdec(segMax[dst * HH + h]);
  float v = expf(scores[t] - m);
  scores[t] = v;
  atomicAdd(&segSum[dst * HH + h], v);
}

// agg[dst, c] += w[e, h(c)] * V[src, c]
__global__ void edge_scatter_kernel(const int* __restrict__ ei, int E,
                                    const float* __restrict__ vSrc,
                                    const float* __restrict__ ex,
                                    const float* __restrict__ segSum,
                                    float* __restrict__ agg) {
  long t = (long)blockIdx.x * blockDim.x + threadIdx.x;
  if (t >= (long)E * CCH) return;
  int e = (int)(t >> 7);
  int c = (int)(t & (CCH - 1));
  int h = c >> 5;
  int src = ei[e], dst = ei[E + e];
  float w = ex[(long)e * HH + h] / segSum[dst * HH + h];
  atomicAdd(&agg[(long)dst * CCH + c], vSrc[(long)src * CCH + c] * w);
}

// ---------------- host orchestration ----------------

extern "C" void kernel_launch(void* const* d_in, const int* in_sizes, int n_in,
                              void* d_out, int out_size, void* d_ws, size_t ws_size,
                              hipStream_t stream) {
  (void)n_in; (void)out_size; (void)ws_size;
  const int N = in_sizes[0] / CCH;
  const int E = in_sizes[2] / 2;

  const float* x_user = (const float*)d_in[0];
  const float* x_item = (const float*)d_in[1];
  const int*   ei_ur  = (const int*)d_in[2];
  const int*   ei_ru  = (const int*)d_in[3];
  const float* Wq_u = (const float*)d_in[4];  const float* bq_u = (const float*)d_in[5];
  const float* Wk_u = (const float*)d_in[6];  const float* bk_u = (const float*)d_in[7];
  const float* Wv_u = (const float*)d_in[8];  const float* bv_u = (const float*)d_in[9];
  const float* Wq_i = (const float*)d_in[10]; const float* bq_i = (const float*)d_in[11];
  const float* Wk_i = (const float*)d_in[12]; const float* bk_i = (const float*)d_in[13];
  const float* Wv_i = (const float*)d_in[14]; const float* bv_i = (const float*)d_in[15];
  const float* Wo = (const float*)d_in[16];   const float* bo = (const float*)d_in[17];
  const float* W1 = (const float*)d_in[18];   const float* b1 = (const float*)d_in[19];
  const float* W2 = (const float*)d_in[20];   const float* b2 = (const float*)d_in[21];
  const float* g1 = (const float*)d_in[22];   const float* be1 = (const float*)d_in[23];
  const float* g2 = (const float*)d_in[24];   const float* be2 = (const float*)d_in[25];
  const float* rel_ur = (const float*)d_in[26]; const float* rel_ru = (const float*)d_in[27];
  const float* tp_ui  = (const float*)d_in[28]; const float* tp_iu  = (const float*)d_in[29];
  const float* in_emb  = (const float*)d_in[30];
  const float* out_emb = (const float*)d_in[31];
  const float* alpha = (const float*)d_in[32];
  const float* beta  = (const float*)d_in[33];

  float* out = (float*)d_out;

  // ---- carve workspace ----
  char* ws = (char*)d_ws;
  size_t off = 0;
  auto alloc = [&](size_t bytes) -> char* {
    off = (off + 255) & ~(size_t)255;
    char* p = ws + off;
    off += bytes;
    return p;
  };

  const long NH = (long)N * HH;
  const long NC = (long)N * CCH;

  // contiguous zero-init region: degrees, segMax x2, segSum x2, agg x2
  const long zwords = 4L * N + 4L * NH + 2L * NC;
  unsigned* zbase = (unsigned*)alloc((size_t)zwords * 4);
  unsigned* in_u  = zbase;
  unsigned* out_u = zbase + N;
  unsigned* in_i  = zbase + 2L * N;
  unsigned* out_i = zbase + 3L * N;
  unsigned* smax_ur = zbase + 4L * N;
  unsigned* smax_ru = zbase + 4L * N + NH;
  float* ssum_ur = (float*)(zbase + 4L * N + 2 * NH);
  float* ssum_ru = (float*)(zbase + 4L * N + 3 * NH);
  float* agg_i = (float*)(zbase + 4L * N + 4 * NH);        // dst of ur edges
  float* agg_u = (float*)(zbase + 4L * N + 4 * NH + NC);   // dst of ru edges

  __bf16* xn_u = (__bf16*)alloc(NC * 2);
  __bf16* xn_i = (__bf16*)alloc(NC * 2);
  float* q_u = (float*)alloc(NC * 4); float* k_u = (float*)alloc(NC * 4); float* v_u = (float*)alloc(NC * 4);
  float* q_i = (float*)alloc(NC * 4); float* k_i = (float*)alloc(NC * 4); float* v_i = (float*)alloc(NC * 4);
  __bf16* WqT_u = (__bf16*)alloc(CCH * CCH * 2); __bf16* WkT_u = (__bf16*)alloc(CCH * CCH * 2);
  __bf16* WvT_u = (__bf16*)alloc(CCH * CCH * 2);
  __bf16* WqT_i = (__bf16*)alloc(CCH * CCH * 2); __bf16* WkT_i = (__bf16*)alloc(CCH * CCH * 2);
  __bf16* WvT_i = (__bf16*)alloc(CCH * CCH * 2);
  __bf16* WoT = (__bf16*)alloc(CCH * CCH * 2);
  __bf16* W1T = (__bf16*)alloc(CCH * 4 * CCH * 2);
  __bf16* W2T = (__bf16*)alloc(4 * CCH * CCH * 2);
  float* sc_ur = (float*)alloc((size_t)E * HH * 4);
  float* sc_ru = (float*)alloc((size_t)E * HH * 4);
  __bf16* aggb_u = (__bf16*)alloc(NC * 2);
  __bf16* aggb_i = (__bf16*)alloc(NC * 2);
  float* y_u = (float*)alloc(NC * 4);
  float* y_i = (float*)alloc(NC * 4);
  __bf16* yn_u = (__bf16*)alloc(NC * 2);
  __bf16* yn_i = (__bf16*)alloc(NC * 2);
  __bf16* h1 = (__bf16*)alloc((size_t)N * 4 * CCH * 2);  // reused user then item (stream-ordered)

  auto blocks1d = [](long n) { return (unsigned)((n + 255) / 256); };

  // 1) zero scratch accumulators
  zero_u32_kernel<<<blocks1d(zwords), 256, 0, stream>>>(zbase, zwords);

  // 2) degrees
  degree_kernel<<<blocks1d(E), 256, 0, stream>>>(ei_ur, E, out_u, in_i);
  degree_kernel<<<blocks1d(E), 256, 0, stream>>>(ei_ru, E, out_i, in_u);

  // 3) weight prep: transpose + bf16
  auto wcvt = [&](const float* W, __bf16* WT, int K, int Nout) {
    wcvt_kernel<<<blocks1d((long)K * Nout), 256, 0, stream>>>(W, WT, K, Nout);
  };
  wcvt(Wq_u, WqT_u, CCH, CCH); wcvt(Wk_u, WkT_u, CCH, CCH); wcvt(Wv_u, WvT_u, CCH, CCH);
  wcvt(Wq_i, WqT_i, CCH, CCH); wcvt(Wk_i, WkT_i, CCH, CCH); wcvt(Wv_i, WvT_i, CCH, CCH);
  wcvt(Wo, WoT, CCH, CCH);
  wcvt(W1, W1T, CCH, 4 * CCH);
  wcvt(W2, W2T, 4 * CCH, CCH);

  // 4) degree-bucket embed + LN1 -> bf16 activations
  embed_ln_kernel<<<N, CCH, 0, stream>>>(x_user, in_u, out_u, in_emb, out_emb, g1, be1, xn_u);
  embed_ln_kernel<<<N, CCH, 0, stream>>>(x_item, in_i, out_i, in_emb, out_emb, g1, be1, xn_i);

  // 5) QKV projections (WMMA bf16, fp32 out)
  auto gemm = [&](const __bf16* A, const __bf16* BT, const float* bias,
                  const float* resid, const unsigned* rowCnt,
                  float* outF, __bf16* outB, int M, int K, int Nout, int mode) {
    dim3 g((unsigned)(M / 16), (unsigned)(Nout / 128));
    gemm16_bf16<<<g, 256, 0, stream>>>(A, BT, bias, resid, rowCnt, outF, outB, M, K, Nout, mode);
  };
  gemm(xn_u, WqT_u, bq_u, nullptr, nullptr, q_u, nullptr, N, CCH, CCH, 0);
  gemm(xn_u, WkT_u, bk_u, nullptr, nullptr, k_u, nullptr, N, CCH, CCH, 0);
  gemm(xn_u, WvT_u, bv_u, nullptr, nullptr, v_u, nullptr, N, CCH, CCH, 0);
  gemm(xn_i, WqT_i, bq_i, nullptr, nullptr, q_i, nullptr, N, CCH, CCH, 0);
  gemm(xn_i, WkT_i, bk_i, nullptr, nullptr, k_i, nullptr, N, CCH, CCH, 0);
  gemm(xn_i, WvT_i, bv_i, nullptr, nullptr, v_i, nullptr, N, CCH, CCH, 0);

  // 6) edge attention passes (segment softmax + weighted V scatter into per-node agg)
  auto edge_pass = [&](const int* ei, const float* qDst, const float* kSrc, const float* vSrc,
                       const unsigned* outS, const unsigned* inD,
                       const float* rel, const float* tp,
                       float* sc, unsigned* smax, float* ssum, float* agg) {
    long th = (long)E * HH;
    edge_score_kernel<<<blocks1d(th), 256, 0, stream>>>(ei, E, qDst, kSrc, outS, inD,
                                                        rel, tp, alpha, beta, sc, smax);
    edge_exp_kernel<<<blocks1d(th), 256, 0, stream>>>(ei, E, sc, smax, ssum);
    long tc = (long)E * CCH;
    edge_scatter_kernel<<<blocks1d(tc), 256, 0, stream>>>(ei, E, vSrc, sc, ssum, agg);
  };
  // user -> item (dst = item)
  edge_pass(ei_ur, q_i, k_u, v_u, out_u, in_i, rel_ur, tp_ui, sc_ur, smax_ur, ssum_ur, agg_i);
  // item -> user (dst = user)
  edge_pass(ei_ru, q_u, k_i, v_i, out_i, in_u, rel_ru, tp_iu, sc_ru, smax_ru, ssum_ru, agg_u);

  // 7) y = x + agg @ Wo + indeg*bo   (Wo pulled through the segment sum)
  cvt_bf16_kernel<<<blocks1d(NC), 256, 0, stream>>>(agg_u, aggb_u, NC);
  cvt_bf16_kernel<<<blocks1d(NC), 256, 0, stream>>>(agg_i, aggb_i, NC);
  gemm(aggb_u, WoT, bo, x_user, in_u, y_u, nullptr, N, CCH, CCH, 2);
  gemm(aggb_i, WoT, bo, x_item, in_i, y_i, nullptr, N, CCH, CCH, 2);

  // 8) LN2 + FFN per type, final residual written straight to d_out
  ln_kernel<<<N, CCH, 0, stream>>>(y_u, g2, be2, yn_u);
  gemm(yn_u, W1T, b1, nullptr, nullptr, nullptr, h1, N, CCH, 4 * CCH, 1);        // gelu -> bf16
  gemm(h1, W2T, b2, y_u, nullptr, out, nullptr, N, 4 * CCH, CCH, 3);             // user output

  ln_kernel<<<N, CCH, 0, stream>>>(y_i, g2, be2, yn_i);
  gemm(yn_i, W1T, b1, nullptr, nullptr, nullptr, h1, N, CCH, 4 * CCH, 1);
  gemm(h1, W2T, b2, y_i, nullptr, out + NC, nullptr, N, 4 * CCH, CCH, 3);        // item output
}
